// MultiHeadSelfAttention_89120571392052
// MI455X (gfx1250) — matrix operand order repaired and checked by reading, not compiled, not run on hardware
//
#include <hip/hip_runtime.h>

#define DMODEL 1024
#define NHEADS 16
#define DHEAD  64

typedef __attribute__((ext_vector_type(16))) __bf16 v16bf;
typedef __attribute__((ext_vector_type(8)))  float  v8f;
typedef __attribute__((ext_vector_type(4)))  int    v4i;

union AFrag { v16bf v; unsigned u[8]; };
union BFrag { v16bf v; uint4 q[2]; };

__device__ __forceinline__ unsigned short f2bf(float f) {
  unsigned u = __float_as_uint(f);
  u += 0x7FFFu + ((u >> 16) & 1u);          // round-to-nearest-even
  return (unsigned short)(u >> 16);
}
__device__ __forceinline__ float bf2f(unsigned short h) {
  return __uint_as_float(((unsigned)h) << 16);
}

// ---------------------------------------------------------------------------
// gfx1250 async global->LDS copy (ASYNCcnt-tracked), with sync fallback.
// ---------------------------------------------------------------------------
#define AS1 __attribute__((address_space(1)))
#define AS3 __attribute__((address_space(3)))

#if defined(__has_builtin)
#if __has_builtin(__builtin_amdgcn_global_load_async_to_lds_b128)
#define HAVE_ASYNC_LDS 1
#endif
#if __has_builtin(__builtin_amdgcn_s_wait_asynccnt)
#define HAVE_WAIT_ASYNC 1
#endif
#endif

__device__ __forceinline__ void ldsCopy16(unsigned short* dst, const unsigned short* src) {
#ifdef HAVE_ASYNC_LDS
  __builtin_amdgcn_global_load_async_to_lds_b128((AS1 v4i*)src, (AS3 v4i*)dst, 0, 0);
#else
  *(uint4*)dst = *(const uint4*)src;
#endif
}
__device__ __forceinline__ void waitAsync() {
#ifdef HAVE_ASYNC_LDS
#ifdef HAVE_WAIT_ASYNC
  __builtin_amdgcn_s_wait_asynccnt(0);
#else
  asm volatile("s_wait_asynccnt 0x0" ::: "memory");
#endif
#endif
}

// ---------------------------------------------------------------------------
// fp32 -> bf16 conversion, vectorized (float4 in, uint2 out)
// ---------------------------------------------------------------------------
__global__ __launch_bounds__(256) void cvt_f32_bf16(const float* __restrict__ src,
                                                    unsigned short* __restrict__ dst,
                                                    int n4) {
  int i = blockIdx.x * blockDim.x + threadIdx.x;
  if (i < n4) {
    float4 f = ((const float4*)src)[i];
    uint2 o;
    o.x = (unsigned)f2bf(f.x) | ((unsigned)f2bf(f.y) << 16);
    o.y = (unsigned)f2bf(f.z) | ((unsigned)f2bf(f.w) << 16);
    ((uint2*)dst)[i] = o;
  }
}

// ---------------------------------------------------------------------------
// Y = X @ W^T  (X: [M x Kd] bf16 row-major, W: [N x Kd] bf16 row-major)
// Output either bf16 (Ybf) or fp32 (Yf). 128x128 block tile, 8 waves,
// each wave computes 32(M) x 64(N) via v_wmma_f32_16x16x32_bf16.
// Double-buffered LDS: next K-tile streams in (async if available) while the
// current tile feeds the matrix pipes; one barrier per K-step.
// ---------------------------------------------------------------------------
#define TK   32
#define XSTR 40   // 32 + 8 pad (bf16 elems); row stride = 80B (16B multiple)

__global__ __launch_bounds__(256) void gemm_xwt(const unsigned short* __restrict__ X,
                                                const unsigned short* __restrict__ W,
                                                unsigned short* __restrict__ Ybf,
                                                float* __restrict__ Yf,
                                                int M, int N, int Kd) {
  __shared__ __align__(16) unsigned short Xs[2][128 * XSTR];
  __shared__ __align__(16) unsigned short Ws[2][128 * XSTR];

  const int tid  = threadIdx.x;
  const int wid  = tid >> 5;
  const int lane = tid & 31;
  const int half = lane >> 4;     // K-half selector per ISA A/B layouts
  const int l15  = lane & 15;
  const int m0 = blockIdx.y * 128, n0 = blockIdx.x * 128;
  const int wm = (wid & 3) * 32;  // 4 waves along M
  const int wn = (wid >> 2) * 64; // 2 waves along N

  // two 16B chunks per thread cover the 128x32 tile
  const int lr0 = tid >> 2, lc0 = (tid & 3) * 8;
  const int lr1 = (tid + 256) >> 2, lc1 = ((tid + 256) & 3) * 8;

  v8f acc[2][4];
#pragma unroll
  for (int mi = 0; mi < 2; ++mi)
#pragma unroll
    for (int ni = 0; ni < 4; ++ni)
#pragma unroll
      for (int r = 0; r < 8; ++r) acc[mi][ni][r] = 0.0f;

  auto loadTile = [&](int buf, int k0) {
    ldsCopy16(&Xs[buf][lr0 * XSTR + lc0], &X[(size_t)(m0 + lr0) * Kd + k0 + lc0]);
    ldsCopy16(&Ws[buf][lr0 * XSTR + lc0], &W[(size_t)(n0 + lr0) * Kd + k0 + lc0]);
    ldsCopy16(&Xs[buf][lr1 * XSTR + lc1], &X[(size_t)(m0 + lr1) * Kd + k0 + lc1]);
    ldsCopy16(&Ws[buf][lr1 * XSTR + lc1], &W[(size_t)(n0 + lr1) * Kd + k0 + lc1]);
  };

  loadTile(0, 0);
  waitAsync();
  __syncthreads();

  const int nk = Kd / TK;
  for (int kt = 0; kt < nk; ++kt) {
    const int buf = kt & 1;
    if (kt + 1 < nk) loadTile(buf ^ 1, (kt + 1) * TK);   // prefetch next tile

    // A fragments (16x32): lane 0-15 K={0..7,16..23}, lane 16-31 K={8..15,24..31}
    AFrag a[2];
#pragma unroll
    for (int mi = 0; mi < 2; ++mi) {
      int base = (wm + mi * 16 + l15) * XSTR + 8 * half;
#pragma unroll
      for (int p = 0; p < 4; ++p) a[mi].u[p]     = *(const unsigned*)&Xs[buf][base + 2 * p];
#pragma unroll
      for (int p = 0; p < 4; ++p) a[mi].u[4 + p] = *(const unsigned*)&Xs[buf][base + 16 + 2 * p];
    }
    // B fragments (32x16): column n = l15, lanes 0-15 K={0..7,16..23}, 16-31 K={8..15,24..31}
    BFrag b[4];
#pragma unroll
    for (int ni = 0; ni < 4; ++ni) {
      int base = (wn + ni * 16 + l15) * XSTR + 8 * half;
      b[ni].q[0] = *(const uint4*)&Ws[buf][base];
      b[ni].q[1] = *(const uint4*)&Ws[buf][base + 16];
    }
#pragma unroll
    for (int mi = 0; mi < 2; ++mi)
#pragma unroll
      for (int ni = 0; ni < 4; ++ni)
        acc[mi][ni] = __builtin_amdgcn_wmma_f32_16x16x32_bf16(
            false, a[mi].v, false, b[ni].v, (short)0, acc[mi][ni], false, false);

    waitAsync();      // this wave's next-tile copies landed
    __syncthreads();  // everyone's landed; current buffer free for reuse
  }

  // epilogue: C VGPR r -> row r + 8*half, col = l15
#pragma unroll
  for (int mi = 0; mi < 2; ++mi)
#pragma unroll
    for (int ni = 0; ni < 4; ++ni)
#pragma unroll
      for (int r = 0; r < 8; ++r) {
        int row = m0 + wm + mi * 16 + r + 8 * half;
        int col = n0 + wn + ni * 16 + l15;
        float v = acc[mi][ni][r];
        if (Yf) Yf[(size_t)row * N + col] = v;
        else    Ybf[(size_t)row * N + col] = f2bf(v);
      }
}

// ---------------------------------------------------------------------------
// RoPE on Q and K in place (bf16 storage, fp32 math). One thread per pair.
// ---------------------------------------------------------------------------
__global__ __launch_bounds__(256) void rope_qk(unsigned short* __restrict__ Q,
                                               unsigned short* __restrict__ K,
                                               const int* __restrict__ pos,
                                               int S, int total) {
  int idx = blockIdx.x * blockDim.x + threadIdx.x;
  if (idx >= total) return;
  int e2 = idx & 511;             // pair index within model dim (= h*32 + i)
  int bs = idx >> 9;              // b*S + s
  int i  = e2 & 31;               // freq index within head
  int s  = bs & (S - 1);
  float p   = (float)pos[s];
  float lf  = -((float)(2 * i) * (1.0f / 64.0f)) * 13.287712379549449f; // log2(1e4)
  float ang = p * exp2f(lf);
  float sn, cs;
  __sincosf(ang, &sn, &cs);
  size_t off = (size_t)bs * DMODEL + (size_t)e2 * 2;
  float q1 = bf2f(Q[off]), q2 = bf2f(Q[off + 1]);
  Q[off]     = f2bf(q1 * cs - q2 * sn);
  Q[off + 1] = f2bf(q1 * sn + q2 * cs);
  float k1 = bf2f(K[off]), k2 = bf2f(K[off + 1]);
  K[off]     = f2bf(k1 * cs - k2 * sn);
  K[off + 1] = f2bf(k1 * sn + k2 * cs);
}

// ---------------------------------------------------------------------------
// Causal flash attention. Block = (b, h, 64-query tile). 8 waves as 4(q)x2(col).
// Q,K,V,O layout: [B*S, 1024] with head h occupying columns [h*64, h*64+64).
// Q and K tiles stream global->LDS via the async path; V is transposed through
// VGPRs into LDS so P@V B-fragments are contiguous ds_load_b128s.
// ---------------------------------------------------------------------------
#define FSTR 72   // 64 + 8 pad (bf16); row stride 144B (16B multiple)
#define SSTR 68   // fp32 score stride

__global__ __launch_bounds__(256) void flash_attn(const unsigned short* __restrict__ Q,
                                                  const unsigned short* __restrict__ K,
                                                  const unsigned short* __restrict__ V,
                                                  unsigned short* __restrict__ O,
                                                  int S) {
  __shared__ __align__(16) unsigned short Qs[64 * FSTR];
  __shared__ __align__(16) unsigned short Ks[64 * FSTR];
  __shared__ __align__(16) unsigned short Vts[64 * FSTR];   // V transposed [d][k]
  __shared__ __align__(16) unsigned short Ps[64 * FSTR];    // softmax probs bf16
  __shared__ float Ss[64 * SSTR];
  __shared__ float mrow[64], lrow[64], arow[64];

  const int tid  = threadIdx.x;
  const int wid  = tid >> 5;
  const int lane = tid & 31;
  const int half = lane >> 4;
  const int l15  = lane & 15;
  const int qt = blockIdx.x, h = blockIdx.y, b = blockIdx.z;
  const int q0 = qt * 64;
  const size_t base = ((size_t)b * S) * DMODEL + (size_t)h * DHEAD;

  // load 64x64 Q tile (async path)
#pragma unroll
  for (int i = 0; i < 2; ++i) {
    int t = tid + i * 256;
    int r = t >> 3, c = (t & 7) * 8;
    ldsCopy16(&Qs[r * FSTR + c], &Q[base + (size_t)(q0 + r) * DMODEL + c]);
  }
  if (tid < 64) { mrow[tid] = -__builtin_inff(); lrow[tid] = 0.0f; }
  waitAsync();
  __syncthreads();

  const int wq = (wid & 3) * 16;   // this wave's 16 query rows
  const int wk = (wid >> 2) * 32;  // this wave's 32-wide column slab (k / d)

  // Q A-fragments for d = [0,32) and [32,64); live across the whole kv loop
  AFrag qa[2];
#pragma unroll
  for (int f = 0; f < 2; ++f) {
    int rb = (wq + l15) * FSTR + f * 32 + 8 * half;
#pragma unroll
    for (int p = 0; p < 4; ++p) qa[f].u[p]     = *(const unsigned*)&Qs[rb + 2 * p];
#pragma unroll
    for (int p = 0; p < 4; ++p) qa[f].u[4 + p] = *(const unsigned*)&Qs[rb + 16 + 2 * p];
  }

  v8f oacc[2];
#pragma unroll
  for (int ni = 0; ni < 2; ++ni)
#pragma unroll
    for (int r = 0; r < 8; ++r) oacc[ni][r] = 0.0f;

  for (int j = 0; j <= qt; ++j) {
    const int k0 = j * 64;
    // K tile via async copy; V tile through VGPRs with on-the-fly transpose
#pragma unroll
    for (int i = 0; i < 2; ++i) {
      int t = tid + i * 256;
      int r = t >> 3, c = (t & 7) * 8;
      ldsCopy16(&Ks[r * FSTR + c], &K[base + (size_t)(k0 + r) * DMODEL + c]);
      uint4 vv = *(const uint4*)&V[base + (size_t)(k0 + r) * DMODEL + c];
      const unsigned short* sp = (const unsigned short*)&vv;
#pragma unroll
      for (int e = 0; e < 8; ++e) Vts[(c + e) * FSTR + r] = sp[e];
    }
    if (j < qt) {  // prefetch next tile into cache (global_prefetch_b8)
      int r = tid >> 3, c = (tid & 7) * 8;
      __builtin_prefetch(&K[base + (size_t)(k0 + 64 + r) * DMODEL + c], 0, 0);
      __builtin_prefetch(&V[base + (size_t)(k0 + 64 + r) * DMODEL + c], 0, 0);
    }
    waitAsync();
    __syncthreads();

    // S = (Q K^T) / sqrt(64), causal-masked on the diagonal tile
#pragma unroll
    for (int ni = 0; ni < 2; ++ni) {
      BFrag kb[2];
#pragma unroll
      for (int f = 0; f < 2; ++f) {
        int bb = (wk + ni * 16 + l15) * FSTR + f * 32 + 8 * half;
        kb[f].q[0] = *(const uint4*)&Ks[bb];
        kb[f].q[1] = *(const uint4*)&Ks[bb + 16];
      }
      v8f s;
#pragma unroll
      for (int r = 0; r < 8; ++r) s[r] = 0.0f;
      s = __builtin_amdgcn_wmma_f32_16x16x32_bf16(false, qa[0].v, false, kb[0].v,
                                                  (short)0, s, false, false);
      s = __builtin_amdgcn_wmma_f32_16x16x32_bf16(false, qa[1].v, false, kb[1].v,
                                                  (short)0, s, false, false);
#pragma unroll
      for (int r = 0; r < 8; ++r) {
        int qr = wq + r + 8 * half;
        int kc = wk + ni * 16 + l15;
        float v = s[r] * 0.125f;
        if (j == qt && kc > qr) v = -__builtin_inff();
        Ss[qr * SSTR + kc] = v;
      }
    }
    __syncthreads();

    // online softmax row update (one thread per query row)
    if (tid < 64) {
      float mold = mrow[tid];
      float mx = mold;
#pragma unroll 8
      for (int c = 0; c < 64; ++c) mx = fmaxf(mx, Ss[tid * SSTR + c]);
      float sum = 0.0f;
#pragma unroll 8
      for (int c = 0; c < 64; ++c) {
        float p = __expf(Ss[tid * SSTR + c] - mx);
        Ps[tid * FSTR + c] = f2bf(p);
        sum += p;
      }
      float alpha = __expf(mold - mx);
      mrow[tid] = mx;
      lrow[tid] = lrow[tid] * alpha + sum;
      arow[tid] = alpha;
    }
    __syncthreads();

    // O = O*alpha + P @ V
    AFrag pa[2];
#pragma unroll
    for (int f = 0; f < 2; ++f) {
      int rb = (wq + l15) * FSTR + f * 32 + 8 * half;
#pragma unroll
      for (int p = 0; p < 4; ++p) pa[f].u[p]     = *(const unsigned*)&Ps[rb + 2 * p];
#pragma unroll
      for (int p = 0; p < 4; ++p) pa[f].u[4 + p] = *(const unsigned*)&Ps[rb + 16 + 2 * p];
    }
#pragma unroll
    for (int ni = 0; ni < 2; ++ni) {
      BFrag vb[2];
#pragma unroll
      for (int f = 0; f < 2; ++f) {
        int bb = (wk + ni * 16 + l15) * FSTR + f * 32 + 8 * half;
        vb[f].q[0] = *(const uint4*)&Vts[bb];
        vb[f].q[1] = *(const uint4*)&Vts[bb + 16];
      }
#pragma unroll
      for (int r = 0; r < 8; ++r) oacc[ni][r] *= arow[wq + r + 8 * half];
      oacc[ni] = __builtin_amdgcn_wmma_f32_16x16x32_bf16(false, pa[0].v, false, vb[0].v,
                                                         (short)0, oacc[ni], false, false);
      oacc[ni] = __builtin_amdgcn_wmma_f32_16x16x32_bf16(false, pa[1].v, false, vb[1].v,
                                                         (short)0, oacc[ni], false, false);
    }
    __syncthreads();
  }

  // normalize and write bf16 attention output
#pragma unroll
  for (int ni = 0; ni < 2; ++ni)
#pragma unroll
    for (int r = 0; r < 8; ++r) {
      int qr = wq + r + 8 * half;
      int dc = wk + ni * 16 + l15;
      float v = oacc[ni][r] / lrow[qr];
      O[base + (size_t)(q0 + qr) * DMODEL + dc] = f2bf(v);
    }
}

// ---------------------------------------------------------------------------
extern "C" void kernel_launch(void* const* d_in, const int* in_sizes, int n_in,
                              void* d_out, int out_size, void* d_ws, size_t ws_size,
                              hipStream_t stream) {
  (void)in_sizes; (void)n_in; (void)out_size; (void)ws_size;
  const float* x  = (const float*)d_in[0];
  const int*  pos = (const int*)d_in[1];
  const float* Wq = (const float*)d_in[2];
  const float* Wk = (const float*)d_in[3];
  const float* Wv = (const float*)d_in[4];
  const float* Wo = (const float*)d_in[5];
  float* out = (float*)d_out;

  const int B = 4, S = 2048, M = B * S;            // M = 8192
  const size_t NX = (size_t)M * DMODEL;            // 8,388,608 elems
  const size_t NW = (size_t)DMODEL * DMODEL;       // 1,048,576 elems

  unsigned short* xb  = (unsigned short*)d_ws;
  unsigned short* wqb = xb  + NX;
  unsigned short* wkb = wqb + NW;
  unsigned short* wvb = wkb + NW;
  unsigned short* wob = wvb + NW;
  unsigned short* Qb  = wob + NW;
  unsigned short* Kb  = Qb  + NX;
  unsigned short* Vb  = Kb  + NX;
  unsigned short* Ab  = Vb  + NX;

  cvt_f32_bf16<<<(int)(NX / 1024), 256, 0, stream>>>(x,  xb,  (int)(NX / 4));
  cvt_f32_bf16<<<(int)(NW / 1024), 256, 0, stream>>>(Wq, wqb, (int)(NW / 4));
  cvt_f32_bf16<<<(int)(NW / 1024), 256, 0, stream>>>(Wk, wkb, (int)(NW / 4));
  cvt_f32_bf16<<<(int)(NW / 1024), 256, 0, stream>>>(Wv, wvb, (int)(NW / 4));
  cvt_f32_bf16<<<(int)(NW / 1024), 256, 0, stream>>>(Wo, wob, (int)(NW / 4));

  dim3 gg(DMODEL / 128, M / 128);   // (8, 64)
  gemm_xwt<<<gg, 256, 0, stream>>>(xb, wqb, Qb, nullptr, M, DMODEL, DMODEL);
  gemm_xwt<<<gg, 256, 0, stream>>>(xb, wkb, Kb, nullptr, M, DMODEL, DMODEL);
  gemm_xwt<<<gg, 256, 0, stream>>>(xb, wvb, Vb, nullptr, M, DMODEL, DMODEL);

  rope_qk<<<(M * 512) / 256, 256, 0, stream>>>(Qb, Kb, pos, S, M * 512);

  dim3 fg(S / 64, NHEADS, B);
  flash_attn<<<fg, 256, 0, stream>>>(Qb, Kb, Vb, Ab, S);

  gemm_xwt<<<gg, 256, 0, stream>>>(Ab, wob, nullptr, out, M, DMODEL, DMODEL);
}
